// MultiHeadAttention_53068615910142
// MI455X (gfx1250) — matrix operand order repaired and checked by reading, not compiled, not run on hardware
//
#include <hip/hip_runtime.h>

typedef __attribute__((ext_vector_type(16))) _Float16 v16h;
typedef __attribute__((ext_vector_type(8)))  _Float16 v8h;
typedef __attribute__((ext_vector_type(8)))  float    v8f;
typedef __attribute__((ext_vector_type(4)))  float    v4f;

#define S_LEN   2048
#define DMODEL  1024
#define NHEAD   16
#define DK      64
#define BATCH   2

static __device__ __forceinline__ v8f wmma_f16(v16h a, v16h b, v8f c) {
  // 8 args: (neg_a, A, neg_b, B, c_mod, C, reuse_a, reuse_b)
  return __builtin_amdgcn_wmma_f32_16x16x32_f16(false, a, false, b, (short)0, c,
                                                false, false);
}

static __device__ __forceinline__ v16h cat16(v8h lo, v8h hi) {
  v16h r;
#pragma unroll
  for (int i = 0; i < 8; ++i) { r[i] = lo[i]; r[8 + i] = hi[i]; }
  return r;
}

// ---------------------------------------------------------------------------
// Elementwise fp32 -> f16 pre-conversion (bandwidth-bound)
// ---------------------------------------------------------------------------
__global__ __launch_bounds__(256)
void cvt_kernel(const float* __restrict__ src, _Float16* __restrict__ dst, int n8) {
  const int i = blockIdx.x * 256 + threadIdx.x;
  if (i < n8) {
    v4f a = *(const v4f*)(src + (size_t)i * 8);
    v4f b = *(const v4f*)(src + (size_t)i * 8 + 4);
    v8h o;
#pragma unroll
    for (int j = 0; j < 4; ++j) { o[j] = (_Float16)a[j]; o[4 + j] = (_Float16)b[j]; }
    *(v8h*)(dst + (size_t)i * 8) = o;
  }
}

// ---------------------------------------------------------------------------
// Projection GEMM (all-f16 operands): out = X @ W.T + bias
// Block = 256 threads (8 waves); each wave computes a 32-row x 64-col strip
// (2 A tiles x 4 B tiles = 8 WMMA per k-step, B tiles shared by both A tiles).
// transposeOut==0 : store f16 [B,H,S,DK] (Q, K) ; ==1 : [B,H,DK,S] (V)
// ---------------------------------------------------------------------------
__global__ __launch_bounds__(256)
void proj_kernel(const _Float16* __restrict__ X, const _Float16* __restrict__ W,
                 const float* __restrict__ bias, _Float16* __restrict__ out,
                 int transposeOut) {
  const int tid  = threadIdx.x;
  const int wave = tid >> 5;
  const int lane = tid & 31;
  const int half = lane >> 4;
  const int n    = lane & 15;

  const int R0 = blockIdx.y * 256 + wave * 32;   // 32 output rows per wave
  const int C0 = blockIdx.x * 64;                // 64 output cols per wave

  v8f acc0[4], acc1[4];
#pragma unroll
  for (int t = 0; t < 4; ++t) { acc0[t] = (v8f){}; acc1[t] = (v8f){}; }

  const _Float16* xr0 = X + (size_t)(R0 + n) * DMODEL;       // A0 row
  const _Float16* xr1 = xr0 + (size_t)16 * DMODEL;           // A1 row
  const _Float16* wr[4];
#pragma unroll
  for (int t = 0; t < 4; ++t)
    wr[t] = W + (size_t)(C0 + t * 16 + n) * DMODEL + half * 16;

  for (int k0 = 0; k0 < DMODEL; k0 += 32) {
    // A tiles: lane<16 holds K {0..15}, lane>=16 {16..31}
    v8h a00 = *(const v8h*)(xr0 + k0 + half * 16);
    v8h a01 = *(const v8h*)(xr0 + k0 + half * 16 + 8);
    v8h a10 = *(const v8h*)(xr1 + k0 + half * 16);
    v8h a11 = *(const v8h*)(xr1 + k0 + half * 16 + 8);
    // B tiles: column n = W row (C0+t*16+n), K = k0 + half*16 + e
    v8h b0l = *(const v8h*)(wr[0] + k0), b0h = *(const v8h*)(wr[0] + k0 + 8);
    v8h b1l = *(const v8h*)(wr[1] + k0), b1h = *(const v8h*)(wr[1] + k0 + 8);
    v8h b2l = *(const v8h*)(wr[2] + k0), b2h = *(const v8h*)(wr[2] + k0 + 8);
    v8h b3l = *(const v8h*)(wr[3] + k0), b3h = *(const v8h*)(wr[3] + k0 + 8);

    const v16h A0 = cat16(a00, a01), A1 = cat16(a10, a11);
    const v16h B0 = cat16(b0l, b0h), B1 = cat16(b1l, b1h);
    const v16h B2 = cat16(b2l, b2h), B3 = cat16(b3l, b3h);

    acc0[0] = wmma_f16(A0, B0, acc0[0]);  acc1[0] = wmma_f16(A1, B0, acc1[0]);
    acc0[1] = wmma_f16(A0, B1, acc0[1]);  acc1[1] = wmma_f16(A1, B1, acc1[1]);
    acc0[2] = wmma_f16(A0, B2, acc0[2]);  acc1[2] = wmma_f16(A1, B2, acc1[2]);
    acc0[3] = wmma_f16(A0, B3, acc0[3]);  acc1[3] = wmma_f16(A1, B3, acc1[3]);
  }

  // bias + store (C layout: lane half selects M 0-7/8-15, col = lane&15)
#pragma unroll
  for (int t = 0; t < 4; ++t) {
    const int c = C0 + t * 16 + n;
    const float bval = bias[c];
    const int h = c >> 6, d = c & 63;
#pragma unroll
    for (int j = 0; j < 8; ++j) {
      const int m  = half * 8 + j;
      const int Ra = R0 + m, Rb = R0 + 16 + m;
      const _Float16 va = (_Float16)(acc0[t][j] + bval);
      const _Float16 vb = (_Float16)(acc1[t][j] + bval);
      const int ba = Ra >> 11, sa = Ra & (S_LEN - 1);
      const int bb = Rb >> 11, sb = Rb & (S_LEN - 1);
      if (!transposeOut) {
        out[((size_t)(ba * NHEAD + h) * S_LEN + sa) * DK + d] = va;
        out[((size_t)(bb * NHEAD + h) * S_LEN + sb) * DK + d] = vb;
      } else {
        out[((size_t)(ba * NHEAD + h) * DK + d) * S_LEN + sa] = va;
        out[((size_t)(bb * NHEAD + h) * DK + d) * S_LEN + sb] = vb;
      }
    }
  }
}

// ---------------------------------------------------------------------------
// One 32-key flash-attention block. Scores computed transposed (S^T = K x Q^T)
// so each lane column is one query; softmax reduction = in-lane + one xor-16.
// MASKED instantiation only used for the single diagonal block per wave.
// ---------------------------------------------------------------------------
template <bool MASKED>
static __device__ __forceinline__ void attn_block(
    int kb, int qg, int half, int n,
    const _Float16* __restrict__ kbase, const _Float16* __restrict__ vbase,
    const v16h& bq0, const v16h& bq1, float& m_i, float& l_i, v8f acc[4]) {
  const v8f zero = {};

  // K tiles as A (M = key row, K-dim = d)
  const _Float16* kp0 = kbase + (size_t)(kb + n) * DK;
  const _Float16* kp1 = kp0 + 16 * DK;
  const v16h ak0l = cat16(*(const v8h*)(kp0 + half * 8),
                          *(const v8h*)(kp0 + 16 + half * 8));
  const v16h ak0h = cat16(*(const v8h*)(kp0 + 32 + half * 8),
                          *(const v8h*)(kp0 + 48 + half * 8));
  const v16h ak1l = cat16(*(const v8h*)(kp1 + half * 8),
                          *(const v8h*)(kp1 + 16 + half * 8));
  const v16h ak1h = cat16(*(const v8h*)(kp1 + 32 + half * 8),
                          *(const v8h*)(kp1 + 48 + half * 8));

  v8f st0 = wmma_f16(ak0l, bq0, zero); st0 = wmma_f16(ak0h, bq1, st0);
  v8f st1 = wmma_f16(ak1l, bq0, zero); st1 = wmma_f16(ak1h, bq1, st1);

  float vmax = -1e30f;
#pragma unroll
  for (int j = 0; j < 8; ++j) {
    if (MASKED) {
      const int key0 = kb + half * 8 + j;
      if (key0 > qg)      st0[j] = -1e9f;
      if (key0 + 16 > qg) st1[j] = -1e9f;
    }
    vmax = fmaxf(vmax, fmaxf(st0[j], st1[j]));
  }
  vmax = fmaxf(vmax, __shfl_xor(vmax, 16, 32));

  const float m_new = fmaxf(m_i, vmax);
  const float corr  = __expf(m_i - m_new);
  float rsum = 0.0f;
#pragma unroll
  for (int j = 0; j < 8; ++j) {
    st0[j] = __expf(st0[j] - m_new);
    st1[j] = __expf(st1[j] - m_new);
    rsum += st0[j] + st1[j];
  }
  rsum += __shfl_xor(rsum, 16, 32);
  l_i = l_i * corr + rsum;
  m_i = m_new;
#pragma unroll
  for (int t = 0; t < 4; ++t)
#pragma unroll
    for (int j = 0; j < 8; ++j) acc[t][j] *= corr;

  // P^T (C layout) -> P (B layout): one xor-16 shuffle per VGPR
  v16h Bp;
#pragma unroll
  for (int j = 0; j < 8; ++j) {
    const float p0 = __shfl_xor(st0[j], 16, 32);
    const float p1 = __shfl_xor(st1[j], 16, 32);
    Bp[j]     = (half == 0) ? (_Float16)st0[j] : (_Float16)p1;
    Bp[8 + j] = (half == 0) ? (_Float16)p0     : (_Float16)st1[j];
  }

  // O^T += V^T x P  (4 d-tiles of 16)
#pragma unroll
  for (int t = 0; t < 4; ++t) {
    const _Float16* vp = vbase + (size_t)(t * 16 + n) * S_LEN + kb;
    const v16h av = cat16(*(const v8h*)(vp + half * 16),
                          *(const v8h*)(vp + half * 16 + 8));
    acc[t] = wmma_f16(av, Bp, acc[t]);
  }
}

__global__ __launch_bounds__(256)
void attn_kernel(const _Float16* __restrict__ qh, const _Float16* __restrict__ kh,
                 const _Float16* __restrict__ vt, float* __restrict__ out) {
  const int tid  = threadIdx.x;
  const int wave = tid >> 5;
  const int lane = tid & 31;
  const int half = lane >> 4;
  const int n    = lane & 15;

  const int bh = blockIdx.y;               // b*16 + h
  const int b  = bh >> 4, h = bh & 15;
  const int qbase = blockIdx.x * 128 + wave * 16;
  const int qg    = qbase + n;             // this lane's query column

  const _Float16* kbase = kh + (size_t)bh * S_LEN * DK;
  const _Float16* vbase = vt + (size_t)bh * DK * S_LEN;

  // Q as B-matrix (Q^T), pre-scaled by 1/sqrt(Dk) = 0.125 (exact in f16)
  const _Float16* qp = qh + ((size_t)bh * S_LEN + qg) * DK;
  v16h bq0 = cat16(*(const v8h*)(qp + half * 8), *(const v8h*)(qp + 16 + half * 8));
  v16h bq1 = cat16(*(const v8h*)(qp + 32 + half * 8),
                   *(const v8h*)(qp + 48 + half * 8));
  {
    const _Float16 sc = (_Float16)0.125f;
#pragma unroll
    for (int i = 0; i < 16; ++i) { bq0[i] *= sc; bq1[i] *= sc; }
  }

  float m_i = -1e30f, l_i = 0.0f;
  v8f acc[4];
#pragma unroll
  for (int t = 0; t < 4; ++t) acc[t] = (v8f){};

  // Unmasked blocks: keys kb..kb+31 all <= min query of this wave.
  const int kfull = ((qbase + 1) >> 5) << 5;
  for (int kb = 0; kb < kfull; kb += 32) {
    __builtin_prefetch(kbase + (size_t)(kb + 32 + n) * DK, 0, 1);
    __builtin_prefetch(vbase + (size_t)n * S_LEN + kb + 32, 0, 1);
    attn_block<false>(kb, qg, half, n, kbase, vbase, bq0, bq1, m_i, l_i, acc);
  }
  // Exactly one diagonal (masked) block: kfull <= qbase < kfull+32, kend=qbase+16.
  attn_block<true>(kfull, qg, half, n, kbase, vbase, bq0, bq1, m_i, l_i, acc);

  // normalize + store: O^T C layout => col = query (lane&15), row = d
  const float inv_l = 1.0f / l_i;
#pragma unroll
  for (int t = 0; t < 4; ++t) {
#pragma unroll
    for (int j = 0; j < 8; ++j) {
      const int d = t * 16 + half * 8 + j;
      out[((size_t)b * S_LEN + qg) * DMODEL + h * 64 + d] = acc[t][j] * inv_l;
    }
  }
}

// ---------------------------------------------------------------------------
extern "C" void kernel_launch(void* const* d_in, const int* in_sizes, int n_in,
                              void* d_out, int out_size, void* d_ws, size_t ws_size,
                              hipStream_t stream) {
  const float* q  = (const float*)d_in[0];
  const float* k  = (const float*)d_in[1];
  const float* v  = (const float*)d_in[2];
  // d_in[3] = causal mask (tril) -- causality applied analytically in-kernel
  const float* Wq = (const float*)d_in[4];
  const float* bq = (const float*)d_in[5];
  const float* Wk = (const float*)d_in[6];
  const float* bk = (const float*)d_in[7];
  const float* Wv = (const float*)d_in[8];
  const float* bv = (const float*)d_in[9];

  const size_t headElems = (size_t)BATCH * NHEAD * S_LEN * DK;  // 4M
  const size_t xElems    = (size_t)BATCH * S_LEN * DMODEL;      // 4M
  const size_t wElems    = (size_t)DMODEL * DMODEL;             // 1M

  _Float16* p   = (_Float16*)d_ws;
  _Float16* qh  = p;  p += headElems;
  _Float16* kh  = p;  p += headElems;
  _Float16* vt  = p;  p += headElems;
  _Float16* xq  = p;  p += xElems;
  _Float16* xk  = p;  p += xElems;
  _Float16* xv  = p;  p += xElems;
  _Float16* wqh = p;  p += wElems;
  _Float16* wkh = p;  p += wElems;
  _Float16* wvh = p;  p += wElems;

  const int nx8 = (int)(xElems / 8), nw8 = (int)(wElems / 8);
  cvt_kernel<<<nx8 / 256, 256, 0, stream>>>(q,  xq,  nx8);
  cvt_kernel<<<nx8 / 256, 256, 0, stream>>>(k,  xk,  nx8);
  cvt_kernel<<<nx8 / 256, 256, 0, stream>>>(v,  xv,  nx8);
  cvt_kernel<<<nw8 / 256, 256, 0, stream>>>(Wq, wqh, nw8);
  cvt_kernel<<<nw8 / 256, 256, 0, stream>>>(Wk, wkh, nw8);
  cvt_kernel<<<nw8 / 256, 256, 0, stream>>>(Wv, wvh, nw8);

  dim3 gp(DMODEL / 64, (BATCH * S_LEN) / 256, 1);   // (16, 16)
  proj_kernel<<<gp, 256, 0, stream>>>(xq, wqh, bq, qh, 0);
  proj_kernel<<<gp, 256, 0, stream>>>(xk, wkh, bk, kh, 0);
  proj_kernel<<<gp, 256, 0, stream>>>(xv, wvh, bv, vt, 1);

  dim3 ga(S_LEN / 128, BATCH * NHEAD, 1);            // (16, 32)
  attn_kernel<<<ga, 256, 0, stream>>>(qh, kh, vt, (float*)d_out);
}
